// GraphAttentionLayer_3899830305187
// MI455X (gfx1250) — matrix operand order repaired and checked by reading, not compiled, not run on hardware
//
#include <hip/hip_runtime.h>
#include <hip/hip_bf16.h>

// v16bf / v8f match the probe-confirmed signature of
// __builtin_amdgcn_wmma_f32_16x16x32_bf16 (bool,v16bf,bool,v16bf,short,v8f,bool,bool)
typedef __attribute__((ext_vector_type(16))) __bf16 v16bf;
typedef __attribute__((ext_vector_type(8)))  float  v8f;
typedef __attribute__((ext_vector_type(4)))  float  v4f;  // clang vector: legal for NT builtins

#define NEG_BIG (-9.0e15f)

static __device__ __forceinline__ __bf16 f2bf(float x) { return (__bf16)x; }

// ---------------------------------------------------------------------------
// Kernel 0: WTb[n][k] = bf16(W[k][n])   (256x256) — makes B fragments for the
// Wh GEMM contiguous along K.
// ---------------------------------------------------------------------------
__global__ void k_prep_wt(const float* __restrict__ W, __bf16* __restrict__ WTb) {
    int n = blockIdx.x;   // 0..255  (output feature)
    int k = threadIdx.x;  // 0..255  (input feature)
    WTb[n * 256 + k] = f2bf(W[k * 256 + n]);
}

// ---------------------------------------------------------------------------
// Kernel 1: Wh = h @ W via v_wmma_f32_16x16x32_bf16.
// One wave per 16-row block; 16 n-tiles => full 16x256 in 16 v8f accumulators.
// Writes WhT bf16 [256][8192] (B operand for the attention GEMM) and
// Wh1/Wh2 = Wh @ a[:F], Wh @ a[F:] reduced through LDS ds_add_f32.
// ---------------------------------------------------------------------------
__global__ void __launch_bounds__(32)
k_phase1(const float* __restrict__ h, const __bf16* __restrict__ WTb,
         const float* __restrict__ a,
         __bf16* __restrict__ WhTb, float* __restrict__ Wh1, float* __restrict__ Wh2) {
    const int lane = threadIdx.x;
    const int m    = lane & 15;   // A row / B column / C column
    const int half = lane >> 4;   // K-group select
    const int r0   = blockIdx.x * 16;

    v8f acc[16];
#pragma unroll
    for (int i = 0; i < 16; ++i) {
        v8f z = {0.f, 0.f, 0.f, 0.f, 0.f, 0.f, 0.f, 0.f};
        acc[i] = z;
    }

    const float* hrow = h + (size_t)(r0 + m) * 256;

    for (int k0 = 0; k0 < 256; k0 += 32) {
        // A fragment (16x32 bf16): lane (m,half); elems 0..15 -> K=k0+half*16+e.
        // One contiguous 16-float run.
        const v4f s0a = *(const v4f*)(hrow + k0 + half * 16);
        const v4f s0b = *(const v4f*)(hrow + k0 + half * 16 + 4);
        const v4f s1a = *(const v4f*)(hrow + k0 + half * 16 + 8);
        const v4f s1b = *(const v4f*)(hrow + k0 + half * 16 + 12);
        v16bf A;
#pragma unroll
        for (int e = 0; e < 4; ++e) {
            A[e]      = f2bf(s0a[e]);
            A[4 + e]  = f2bf(s0b[e]);
            A[8 + e]  = f2bf(s1a[e]);
            A[12 + e] = f2bf(s1b[e]);
        }

#pragma unroll
        for (int nt = 0; nt < 16; ++nt) {
            // B fragment (32x16): lane column N = nt*16+m, K = k0+half*16+e,
            // contiguous 16 bf16 (32B aligned) in WTb.
            v16bf B = *(const v16bf*)(WTb + (size_t)(nt * 16 + m) * 256 + k0 + half * 16);
            acc[nt] = __builtin_amdgcn_wmma_f32_16x16x32_bf16(
                false, A, false, B, (short)0, acc[nt], false, false);
        }
    }

    // D layout: VGPR v, lane -> row M = v + 8*half, col N = nt*16 + m.
    __shared__ float lw1[16], lw2[16];
    if (lane < 16) { lw1[lane] = 0.f; lw2[lane] = 0.f; }
    __syncthreads();

    float p1[8], p2[8];
#pragma unroll
    for (int v = 0; v < 8; ++v) { p1[v] = 0.f; p2[v] = 0.f; }

#pragma unroll
    for (int nt = 0; nt < 16; ++nt) {
        const int   n  = nt * 16 + m;
        const float a1 = a[n];
        const float a2 = a[256 + n];
        alignas(16) __bf16 row8[8];
#pragma unroll
        for (int v = 0; v < 8; ++v) {
            float f = acc[nt][v];
            row8[v] = f2bf(f);
            p1[v] += f * a1;
            p2[v] += f * a2;
        }
        // WhT[n][r0 + 8*half .. +7]: 8 bf16 = one 16B store
        *(uint4*)(WhTb + (size_t)n * 8192 + r0 + half * 8) = *(uint4*)row8;
    }
#pragma unroll
    for (int v = 0; v < 8; ++v) {
        atomicAdd(&lw1[v + 8 * half], p1[v]);
        atomicAdd(&lw2[v + 8 * half], p2[v]);
    }
    __syncthreads();
    if (lane < 16) {
        Wh1[r0 + lane] = lw1[lane];
        Wh2[r0 + lane] = lw2[lane];
    }
}

// ---------------------------------------------------------------------------
// Kernel 2: gmax = max_j Wh2[j]  (upper-bound softmax stabilizer)
// ---------------------------------------------------------------------------
__global__ void k_gmax(const float* __restrict__ Wh2, float* __restrict__ gmax) {
    __shared__ float sm[256];
    float mx = -3.4e38f;
    for (int j = threadIdx.x; j < 8192; j += 256) mx = fmaxf(mx, Wh2[j]);
    sm[threadIdx.x] = mx;
    __syncthreads();
    for (int s = 128; s > 0; s >>= 1) {
        if (threadIdx.x < s) sm[threadIdx.x] = fmaxf(sm[threadIdx.x], sm[threadIdx.x + s]);
        __syncthreads();
    }
    if (threadIdx.x == 0) gmax[0] = sm[0];
}

// ---------------------------------------------------------------------------
// Kernel 3: fused masked softmax + attention @ Wh + ELU.
// One wave per 16-row block; streams adj once with NON-TEMPORAL loads (256MB
// single-use stream must not evict the L2-resident 4MB WhT / 32KB Wh2),
// builds P tiles in the A-fragment layout, accumulates P@Wh with bf16 WMMA.
// stab = leaky(Wh1[i] + max_j Wh2[j]) >= true row max, so exp() never
// overflows and the softmax result is mathematically unchanged.
// ---------------------------------------------------------------------------
__global__ void __launch_bounds__(32)
k_attn(const float* __restrict__ adj, const __bf16* __restrict__ WhTb,
       const float* __restrict__ Wh1, const float* __restrict__ Wh2,
       const float* __restrict__ gmaxp, float* __restrict__ out) {
    const int lane = threadIdx.x;
    const int m    = lane & 15;
    const int half = lane >> 4;
    const int r0   = blockIdx.x * 16;

    const float wh1  = Wh1[r0 + m];
    const float gmax = gmaxp[0];
    const float t    = wh1 + gmax;
    const float stab = t > 0.f ? t : 0.2f * t;  // leaky_relu of the bound

    v8f acc[16];
#pragma unroll
    for (int i = 0; i < 16; ++i) {
        v8f z = {0.f, 0.f, 0.f, 0.f, 0.f, 0.f, 0.f, 0.f};
        acc[i] = z;
    }
    float lsum = 0.f;

    const float* adjrow = adj + (size_t)(r0 + m) * 8192;

    for (int j0 = 0; j0 < 8192; j0 += 32) {
        v16bf A;
        // segment 0: elems 0..7 -> j = j0 + half*16 + e
        {
            const int base = j0 + half * 16;
            const v4f aa = __builtin_nontemporal_load((const v4f*)(adjrow + base));
            const v4f ab = __builtin_nontemporal_load((const v4f*)(adjrow + base + 4));
            const v4f wa = *(const v4f*)(Wh2 + base);
            const v4f wb = *(const v4f*)(Wh2 + base + 4);
#pragma unroll
            for (int e = 0; e < 8; ++e) {
                float adjv = (e < 4) ? aa[e] : ab[e - 4];
                float wh2v = (e < 4) ? wa[e] : wb[e - 4];
                float ev = wh1 + wh2v;
                float le = ev > 0.f ? ev : 0.2f * ev;
                float me = adjv > 0.f ? le : NEG_BIG;
                float p  = __expf(me - stab);
                lsum += p;
                A[e] = f2bf(p);
            }
        }
        // segment 1: elems 8..15 -> j = j0 + half*16 + 8 + (e-8)
        {
            const int base = j0 + half * 16 + 8;
            const v4f aa = __builtin_nontemporal_load((const v4f*)(adjrow + base));
            const v4f ab = __builtin_nontemporal_load((const v4f*)(adjrow + base + 4));
            const v4f wa = *(const v4f*)(Wh2 + base);
            const v4f wb = *(const v4f*)(Wh2 + base + 4);
#pragma unroll
            for (int e = 0; e < 8; ++e) {
                float adjv = (e < 4) ? aa[e] : ab[e - 4];
                float wh2v = (e < 4) ? wa[e] : wb[e - 4];
                float ev = wh1 + wh2v;
                float le = ev > 0.f ? ev : 0.2f * ev;
                float me = adjv > 0.f ? le : NEG_BIG;
                float p  = __expf(me - stab);
                lsum += p;
                A[8 + e] = f2bf(p);
            }
        }
#pragma unroll
        for (int nt = 0; nt < 16; ++nt) {
            // B = Wh[j0+half*16+e][nt*16+m] = WhT[nt*16+m][j0+half*16+e]
            v16bf B = *(const v16bf*)(WhTb + (size_t)(nt * 16 + m) * 8192 + j0 + half * 16);
            acc[nt] = __builtin_amdgcn_wmma_f32_16x16x32_bf16(
                false, A, false, B, (short)0, acc[nt], false, false);
        }
    }

    // Row sums: lanes (m) and (m+16) hold disjoint K-halves of row m.
    __shared__ float ls[16];
    if (lane < 16) ls[lane] = 0.f;
    __syncthreads();
    atomicAdd(&ls[m], lsum);
    __syncthreads();

    float inv[8];
#pragma unroll
    for (int v = 0; v < 8; ++v) inv[v] = 1.0f / ls[v + 8 * half];

#pragma unroll
    for (int nt = 0; nt < 16; ++nt) {
#pragma unroll
        for (int v = 0; v < 8; ++v) {
            const int   row = r0 + v + 8 * half;
            const float val = acc[nt][v] * inv[v];
            const float o   = val > 0.f ? val : (__expf(val) - 1.0f);  // ELU(alpha=1)
            __builtin_nontemporal_store(o, &out[(size_t)row * 256 + nt * 16 + m]);
        }
    }
}

// ---------------------------------------------------------------------------
// Launch: prep WT -> Wh GEMM (+Wh1/Wh2) -> gmax -> fused attention.
// Workspace: 4MB WhT bf16 + 32KB Wh1 + 32KB Wh2 + gmax + 128KB WT bf16.
// ---------------------------------------------------------------------------
extern "C" void kernel_launch(void* const* d_in, const int* in_sizes, int n_in,
                              void* d_out, int out_size, void* d_ws, size_t ws_size,
                              hipStream_t stream) {
    const float* h   = (const float*)d_in[0];
    const float* adj = (const float*)d_in[1];
    const float* W   = (const float*)d_in[2];
    const float* a   = (const float*)d_in[3];
    float*       out = (float*)d_out;

    char*    ws   = (char*)d_ws;
    __bf16*  WhTb = (__bf16*)ws;                                  // 256*8192*2 = 4 MB
    float*   Wh1  = (float*)(ws + 4u * 1024u * 1024u);            // 32 KB
    float*   Wh2  = (float*)(ws + 4u * 1024u * 1024u + 32u * 1024u);
    float*   gmx  = (float*)(ws + 4u * 1024u * 1024u + 64u * 1024u);
    __bf16*  WTb  = (__bf16*)(ws + 4u * 1024u * 1024u + 64u * 1024u + 256u);

    hipLaunchKernelGGL(k_prep_wt, dim3(256), dim3(256), 0, stream, W, WTb);
    hipLaunchKernelGGL(k_phase1,  dim3(512), dim3(32),  0, stream, h, WTb, a, WhTb, Wh1, Wh2);
    hipLaunchKernelGGL(k_gmax,    dim3(1),   dim3(256), 0, stream, Wh2, gmx);
    hipLaunchKernelGGL(k_attn,    dim3(512), dim3(32),  0, stream, adj, WhTb, Wh1, Wh2, gmx, out);
}
